// RNNBase_49555332662170
// MI455X (gfx1250) — matrix operand order repaired and checked by reading, not compiled, not run on hardware
//
#include <hip/hip_runtime.h>
#include <hip/hip_bf16.h>
#include <math.h>

// ---------------------------------------------------------------------------
// Fused multi-instance tanh RNN (HFTA style), CDNA5 / gfx1250, wave32 + WMMA.
//   B=8 instances, T=512, N=64 batch, I=H=256, L=2 layers.
// Phase 1 (proj):  xin[b,t,n,h] = x[b,t,n,:] @ w_ih[b]^T + b_ih + b_hh  (WMMA)
// Phase 2 (rnn) :  h = tanh(xin[t] + h @ w_hh[b]^T), W_hh^T resident in LDS.
// LDS rows padded to 264 halfs (132 dwords) so lane-strided ds_load_b128s
// step 4 banks instead of 0 (kills the 16-way conflicts of a 256 stride).
// ---------------------------------------------------------------------------

typedef __attribute__((ext_vector_type(16))) __bf16 bf16x16;
typedef __attribute__((ext_vector_type(8)))  __bf16 bf16x8;
typedef __attribute__((ext_vector_type(8)))  float  f32x8;

namespace {
constexpr int kB = 8, kT = 512, kN = 64, kH = 256, kL = 2;
constexpr int kLdsW = kH + 8;                       // padded LDS row stride (halfs)
constexpr int kWHalfs = kH * kLdsW;                 // W^T tile: 67584 halfs
constexpr int kHHalfs = kN * kLdsW;                 // hidden buf: 16896 halfs
constexpr size_t kWBytes = (size_t)kWHalfs * 2;     // 135168
constexpr size_t kHBytes = (size_t)kHHalfs * 2;     //  33792
constexpr size_t kSeqElems = (size_t)kT * kN * kH;  // per instance
constexpr size_t kAllElems = (size_t)kB * kSeqElems;
}

union AFrag { bf16x16 v; bf16x8 h[2]; };

__device__ __forceinline__ float fast_tanh(float x) {
#if defined(__AMDGCN__)
#if __has_builtin(__builtin_amdgcn_tanhf)
  return __builtin_amdgcn_tanhf(x);               // v_tanh_f32 (CDNA5 TRANS op)
#elif __has_builtin(__builtin_amdgcn_tanh_f32)
  return __builtin_amdgcn_tanh_f32(x);
#else
  float r;
  asm("v_tanh_f32 %0, %1" : "=v"(r) : "v"(x));
  return r;
#endif
#else
  return tanhf(x);
#endif
}

// Per-wave 64x256 GEMM slice: D += A[64x256] * W^T, bf16 LDS in, f32 acc.
// Wave owns output columns [n_base, n_base+32). 64 WMMAs per call.
__device__ __forceinline__ void mm64x256(const __bf16* A,    // [64][kLdsW] LDS
                                         const __bf16* BT,   // [256][kLdsW], BT[n][k]=W[n][k]
                                         int lane, int n_base,
                                         f32x8 acc[4][2]) {
  const int lsel = lane >> 4;
  const int lrow = lane & 15;
#pragma unroll
  for (int ks = 0; ks < 8; ++ks) {
    const int k0 = ks * 32;
    // B fragment: lane holds column n = n_base + lrow (+16), K k0+8*lsel+[0..7] & k0+16+8*lsel+[0..7].
    const int nb0 = (n_base + lrow) * kLdsW + k0 + lsel * 8;
    const int nb1 = nb0 + 16 * kLdsW;
    AFrag b0, b1;
    b0.h[0] = *(const bf16x8*)(BT + nb0);
    b0.h[1] = *(const bf16x8*)(BT + nb0 + 16);
    b1.h[0] = *(const bf16x8*)(BT + nb1);
    b1.h[1] = *(const bf16x8*)(BT + nb1 + 16);
#pragma unroll
    for (int mt = 0; mt < 4; ++mt) {
      const int row = mt * 16 + lrow;
      // A per ISA: lanes 0-15 hold K k0+[0..7] & k0+[16..23],
      //            lanes 16-31 hold K k0+[8..15] & k0+[24..31].
      AFrag a;
      a.h[0] = *(const bf16x8*)(A + row * kLdsW + k0 + lsel * 8);
      a.h[1] = *(const bf16x8*)(A + row * kLdsW + k0 + 16 + lsel * 8);
      acc[mt][0] = __builtin_amdgcn_wmma_f32_16x16x32_bf16(
          false, a.v, false, b0.v, (short)0, acc[mt][0], false, false);
      acc[mt][1] = __builtin_amdgcn_wmma_f32_16x16x32_bf16(
          false, a.v, false, b1.v, (short)0, acc[mt][1], false, false);
    }
  }
}

// ---------------------------------------------------------------------------
// Phase 1: input projection for one layer. grid = B*(T*N/64) = 4096, 256 thr.
// ---------------------------------------------------------------------------
__global__ void __launch_bounds__(256) rnn_proj_kernel(
    const float* __restrict__ X,    // [B, T*N, 256]
    const float* __restrict__ Wih,  // [B, 256, 256]
    const float* __restrict__ bih,  // [B, 256]
    const float* __restrict__ bhh,  // [B, 256]
    float* __restrict__ xin) {      // [B, T*N, 256]
  const int blk = blockIdx.x;
  const int b  = blk >> 9;
  const int r0 = (blk & 511) * 64;
  extern __shared__ char smem[];
  __bf16* wT = (__bf16*)smem;                  // padded W^T in bf16
  __bf16* xt = (__bf16*)(smem + kWBytes);      // padded 64x256 x tile in bf16
  const int tid = threadIdx.x;

  const float* wb = Wih + (size_t)b * kH * kH;
  for (int i = tid; i < kH * kH; i += 256) {
    const int r = i >> 8, c = i & 255;
    wT[r * kLdsW + c] = (__bf16)wb[i];         // store W[n][k], k contiguous
  }
  const float* xb = X + ((size_t)b * (kT * kN) + r0) * kH;
  for (int i = tid; i < kN * kH; i += 256)
    xt[(i >> 8) * kLdsW + (i & 255)] = (__bf16)__builtin_nontemporal_load(xb + i);
  __syncthreads();

  const int lane = tid & 31, wave = tid >> 5;
  const int n_base = wave * 32, lsel = lane >> 4, lrow = lane & 15;
  f32x8 acc[4][2];
#pragma unroll
  for (int mt = 0; mt < 4; ++mt) { acc[mt][0] = 0.f; acc[mt][1] = 0.f; }

  // Biases folded per column (fixed per lane/nt).
  const float* bi = bih + (size_t)b * kH;
  const float* bh = bhh + (size_t)b * kH;
  float bias[2];
#pragma unroll
  for (int nt = 0; nt < 2; ++nt) {
    const int col = n_base + nt * 16 + lrow;
    bias[nt] = bi[col] + bh[col];
  }

  mm64x256(xt, wT, lane, n_base, acc);

  float* o = xin + ((size_t)b * (kT * kN) + r0) * kH;
#pragma unroll
  for (int mt = 0; mt < 4; ++mt)
#pragma unroll
    for (int nt = 0; nt < 2; ++nt) {
      const int col = n_base + nt * 16 + lrow;
#pragma unroll
      for (int r = 0; r < 8; ++r) {
        const int row = mt * 16 + r + lsel * 8;
        __builtin_nontemporal_store(acc[mt][nt][r] + bias[nt], o + row * kH + col);
      }
    }
}

// ---------------------------------------------------------------------------
// Phase 2: persistent recurrence for one layer. grid = B, 256 threads.
//   W_hh^T resident in LDS (bf16); hidden state ping-pong. Per step:
//   512 v_wmma_f32_16x16x32_bf16 per WG, xin loads overlapped, hw tanh.
// ---------------------------------------------------------------------------
__global__ void __launch_bounds__(256) rnn_recur_kernel(
    const float* __restrict__ xin,  // [B, T*N, 256] (biases folded in)
    const float* __restrict__ Whh,  // [B, 256, 256] (this layer)
    const float* __restrict__ hx,   // [B, L, N, 256]
    float* __restrict__ y,          // [B, T*N, 256]
    float* __restrict__ hn,         // [B, L, N, 256]
    int layer) {
  const int b = blockIdx.x;
  extern __shared__ char smem[];
  __bf16* wT = (__bf16*)smem;
  __bf16* h0 = (__bf16*)(smem + kWBytes);
  __bf16* h1 = h0 + kHHalfs;
  const int tid = threadIdx.x;

  const float* wb = Whh + (size_t)b * kH * kH;
  for (int i = tid; i < kH * kH; i += 256) {
    const int r = i >> 8, c = i & 255;
    wT[r * kLdsW + c] = (__bf16)wb[i];                // W_hh[n][k] in bf16
  }
  const float* hxb = hx + ((size_t)b * kL + layer) * kN * kH;
  for (int i = tid; i < kN * kH; i += 256)
    h0[(i >> 8) * kLdsW + (i & 255)] = (__bf16)hxb[i];
  __syncthreads();

  const int lane = tid & 31, wave = tid >> 5;
  const int n_base = wave * 32, lsel = lane >> 4, lrow = lane & 15;
  const float* xin_b = xin + (size_t)b * kSeqElems;
  float*       y_b   = y   + (size_t)b * kSeqElems;
  float*       hn_b  = hn  + ((size_t)b * kL + layer) * kN * kH;

  for (int t = 0; t < kT; ++t) {
    const __bf16* hc  = (t & 1) ? h1 : h0;
    __bf16*       hnx = (t & 1) ? h0 : h1;
    const float* xin_t = xin_b + (size_t)t * kN * kH;
    float*       y_t   = y_b   + (size_t)t * kN * kH;

    // Issue this step's xin loads up-front; they overlap the WMMA chain.
    float xv[4][2][8];
#pragma unroll
    for (int mt = 0; mt < 4; ++mt)
#pragma unroll
      for (int nt = 0; nt < 2; ++nt) {
        const int col = n_base + nt * 16 + lrow;
#pragma unroll
        for (int r = 0; r < 8; ++r) {
          const int row = mt * 16 + r + lsel * 8;
          xv[mt][nt][r] = xin_t[row * kH + col];
        }
      }

    // Pull next step's xin slab toward cache while WMMAs run (256B/thread).
    if (t + 1 < kT)
      __builtin_prefetch(xin_t + (size_t)kN * kH + (size_t)tid * 64, 0, 1);

    f32x8 acc[4][2];
#pragma unroll
    for (int mt = 0; mt < 4; ++mt) { acc[mt][0] = 0.f; acc[mt][1] = 0.f; }

    mm64x256(hc, wT, lane, n_base, acc);

#pragma unroll
    for (int mt = 0; mt < 4; ++mt)
#pragma unroll
      for (int nt = 0; nt < 2; ++nt) {
        const int col = n_base + nt * 16 + lrow;
#pragma unroll
        for (int r = 0; r < 8; ++r) {
          const int row = mt * 16 + r + lsel * 8;
          const float v = fast_tanh(acc[mt][nt][r] + xv[mt][nt][r]);
          __builtin_nontemporal_store(v, y_t + row * kH + col);
          hnx[row * kLdsW + col] = (__bf16)v;
          if (t == kT - 1) hn_b[row * kH + col] = v;
        }
      }
    __syncthreads();  // all waves done with hc & hnx before roles swap
  }
}

// ---------------------------------------------------------------------------
extern "C" void kernel_launch(void* const* d_in, const int* in_sizes, int n_in,
                              void* d_out, int out_size, void* d_ws, size_t ws_size,
                              hipStream_t stream) {
  const float* x    = (const float*)d_in[0];
  const float* hx   = (const float*)d_in[1];
  const float* wih0 = (const float*)d_in[2];
  const float* whh0 = (const float*)d_in[3];
  const float* bih0 = (const float*)d_in[4];
  const float* bhh0 = (const float*)d_in[5];
  const float* wih1 = (const float*)d_in[6];
  const float* whh1 = (const float*)d_in[7];
  const float* bih1 = (const float*)d_in[8];
  const float* bhh1 = (const float*)d_in[9];

  float* xin = (float*)d_ws;               // [B,T,N,H] 256 MB
  float* y0  = xin + kAllElems;            // layer-0 outputs, 256 MB
  float* out = (float*)d_out;              // [B,T,N,H]
  float* hn  = out + kAllElems;            // [B,L,N,H]

  const size_t projLds = kWBytes + kHBytes;       // 165 KB
  const size_t rnnLds  = kWBytes + 2 * kHBytes;   // 198 KB (< 320 KB WGP LDS)
  (void)hipFuncSetAttribute((const void*)rnn_proj_kernel,
      hipFuncAttributeMaxDynamicSharedMemorySize, (int)projLds);
  (void)hipFuncSetAttribute((const void*)rnn_recur_kernel,
      hipFuncAttributeMaxDynamicSharedMemorySize, (int)rnnLds);

  const int projGrid = kB * (kT * kN / 64);  // 4096

  // Layer 0
  rnn_proj_kernel<<<projGrid, 256, projLds, stream>>>(x, wih0, bih0, bhh0, xin);
  rnn_recur_kernel<<<kB, 256, rnnLds, stream>>>(xin, whh0, hx, y0, hn, 0);
  // Layer 1 (xin buffer reused)
  rnn_proj_kernel<<<projGrid, 256, projLds, stream>>>(y0, wih1, bih1, bhh1, xin);
  rnn_recur_kernel<<<kB, 256, rnnLds, stream>>>(xin, whh1, hx, out, hn, 1);
}
